// RNN_79396765434195
// MI455X (gfx1250) — matrix operand order repaired and checked by reading, not compiled, not run on hardware
//
#include <hip/hip_runtime.h>
#include <hip/hip_bf16.h>

#define H     128
#define LAG   5
#define STEPS 10
#define BATCH 65536
#define FH    5          // forecast horizon
#define G4H   512        // 4*H (gate dim)
#define KCAT  256        // concatenated K ( [x | h] )
#define BT    32         // batch rows per workgroup
#define AST   264        // LDS activation row stride (elements, 16B-aligned rows)
#define WPK_L 131072     // packed-weight elements per layer (256*512)

typedef __attribute__((ext_vector_type(16))) __bf16 v16bf;
typedef __attribute__((ext_vector_type(8)))  float  v8f;

union FragAB { uint4 q[2]; v16bf v; };

__device__ __forceinline__ unsigned short f2bf(float x) {
    unsigned int u = __float_as_uint(x);
    u += 0x7FFFu + ((u >> 16) & 1u);          // round-to-nearest-even
    return (unsigned short)(u >> 16);
}
__device__ __forceinline__ float bf2f(unsigned short h) {
    return __uint_as_float(((unsigned int)h) << 16);
}
__device__ __forceinline__ float sigmoidf_(float x) { return 1.0f / (1.0f + __expf(-x)); }
__device__ __forceinline__ float tanhf_(float x)    { return 1.0f - 2.0f / (__expf(2.0f * x) + 1.0f); }

// Optimization barrier: stops LICM from hoisting the (loop-invariant) weight
// loads out of the timestep loop, which blew the VGPR budget and spilled the
// whole B-matrix to scratch. Weights must be re-read from L2 each iteration.
__device__ __forceinline__ void opt_barrier() { asm volatile("" ::: "memory"); }

// ---------------------------------------------------------------------------
// Pack fp32 weights into bf16 W^T, concat layout [K=256][N=512] per layer:
// rows 0..127 = W_ih^T, rows 128..255 = W_hh^T  (so B-fragment = 32B/lane).
// ---------------------------------------------------------------------------
__global__ void pack_weights(const float* __restrict__ Wih0, const float* __restrict__ Whh0,
                             const float* __restrict__ Wih1, const float* __restrict__ Whh1,
                             unsigned short* __restrict__ wpack) {
    int gid = blockIdx.x * blockDim.x + threadIdx.x;   // 0 .. 2*WPK_L-1
    int L   = gid >> 17;
    int rem = gid & (WPK_L - 1);
    int k   = rem >> 9;        // 0..255
    int g   = rem & 511;       // 0..511
    const float* Wih = L ? Wih1 : Wih0;
    const float* Whh = L ? Whh1 : Whh0;
    float v = (k < H) ? Wih[g * H + k] : Whh[g * H + (k - H)];
    wpack[L * WPK_L + k * G4H + g] = f2bf(v);
}

// ---------------------------------------------------------------------------
// One LSTM cell for a 32x128 batch tile:
//   gates[32,512] = act[32,256] @ Wcat[256,512]  (bf16 WMMA, f32 acc)
// This wave: m-tile `mt` (16 rows), hidden cols [hcBase, hcBase+32), all 4 gates.
// ---------------------------------------------------------------------------
__device__ __forceinline__ void lstm_layer(const unsigned short (*act)[AST],
                                           const unsigned short* __restrict__ wL,
                                           const float* __restrict__ biasRow,
                                           int mt, int hcBase, int lane,
                                           v8f cst[2], v8f hout[2]) {
    const int hi   = lane >> 4;
    const int ln16 = lane & 15;

    // A fragments (ISA 16-bit A layout: lane = M, K halves selected by lane/16)
    FragAB a[8];
#pragma unroll
    for (int kc = 0; kc < 8; ++kc) {
        const int row = mt * 16 + ln16;
        const int k0  = kc * 32 + 8 * hi;
        a[kc].q[0] = *(const uint4*)&act[row][k0];
        a[kc].q[1] = *(const uint4*)&act[row][k0 + 16];
    }

    v8f acc[4][2];
#pragma unroll
    for (int Gg = 0; Gg < 4; ++Gg) {
#pragma unroll
        for (int t = 0; t < 2; ++t) {
            const int colBase = Gg * 128 + hcBase + t * 16;
            // Bound in-flight weight loads to one tile's worth (keeps VGPR
            // pressure under the spill threshold).
            opt_barrier();
            v8f c;
#pragma unroll
            for (int e = 0; e < 8; ++e) c[e] = 0.0f;
#pragma unroll
            for (int kc = 0; kc < 8; ++kc) {
                // B fragment: lane l -> N col (l%16), K in table order (8*hi + s, 16 + 8*hi + s-8)
                FragAB b;
                const unsigned short* p = wL + (kc * 32 + 8 * hi) * G4H + colBase + ln16;
#pragma unroll
                for (int s2 = 0; s2 < 16; ++s2)
                    b.v[s2] = __builtin_bit_cast(__bf16, p[((s2 & 7) + ((s2 >> 3) << 4)) * G4H]);
                c = __builtin_amdgcn_wmma_f32_16x16x32_bf16(
                        false, a[kc].v, false, b.v, (short)0, c, false, false);
            }
            const float bv = biasRow[colBase + ln16];   // C: N = lane%16
#pragma unroll
            for (int e = 0; e < 8; ++e) c[e] += bv;
            acc[Gg][t] = c;
        }
    }

    // Pointwise LSTM update, fully in registers (fragment layouts line up).
#pragma unroll
    for (int t = 0; t < 2; ++t) {
        v8f gi = acc[0][t], gf = acc[1][t], gg = acc[2][t], go = acc[3][t];
        v8f cn, hh;
#pragma unroll
        for (int e = 0; e < 8; ++e) {
            float cv = sigmoidf_(gf[e]) * cst[t][e] + sigmoidf_(gi[e]) * tanhf_(gg[e]);
            cn[e] = cv;
            hh[e] = sigmoidf_(go[e]) * tanhf_(cv);
        }
        cst[t]  = cn;
        hout[t] = hh;
    }
}

// ---------------------------------------------------------------------------
// Main: 2-layer LSTM over 10 steps for a 32-row batch tile per workgroup.
// ---------------------------------------------------------------------------
__global__ __launch_bounds__(256)
void lstm2_forward(const float* __restrict__ inputs,
                   const float* __restrict__ W_inp, const float* __restrict__ b_inp,
                   const float* __restrict__ b_ih0, const float* __restrict__ b_hh0,
                   const float* __restrict__ b_ih1, const float* __restrict__ b_hh1,
                   const float* __restrict__ W_out, const float* __restrict__ b_out,
                   const float* __restrict__ W_fc,  const float* __restrict__ b_fc,
                   const unsigned short* __restrict__ wpack,
                   float* __restrict__ out) {
    __shared__ __align__(16) unsigned short act0[BT][AST];  // [ xp | h1_prev ]
    __shared__ __align__(16) unsigned short act1[BT][AST];  // [ h1 | h2_prev ]
    __shared__ float biasL[2][G4H];

    const int tid    = threadIdx.x;
    const int lane   = tid & 31;
    const int w      = tid >> 5;
    const int mt     = w & 1;        // m-tile (16 batch rows)
    const int hcBase = (w >> 1) * 32;
    const int hi     = lane >> 4;
    const int ln16   = lane & 15;
    const int bBase  = blockIdx.x * BT;

    for (int n = tid; n < G4H; n += 256) {
        biasL[0][n] = b_ih0[n] + b_hh0[n];
        biasL[1][n] = b_ih1[n] + b_hh1[n];
    }
    for (int idx = tid; idx < BT * H; idx += 256) {
        int r = idx >> 7, c = idx & 127;
        act0[r][H + c] = 0;   // h1 = 0
        act1[r][c]     = 0;
        act1[r][H + c] = 0;   // h2 = 0
    }

    v8f c1[2], c2[2], h1f[2], h2f[2];
#pragma unroll
    for (int t = 0; t < 2; ++t)
#pragma unroll
        for (int e = 0; e < 8; ++e) { c1[t][e] = 0.0f; c2[t][e] = 0.0f; }

    __syncthreads();

    for (int s = 0; s < STEPS; ++s) {
        // Stop cross-iteration hoisting of weight loads (see opt_barrier()).
        opt_barrier();

        // ---- input projection: xp = x @ W_inp^T + b_inp  -> act0[:, 0:128]
        {
            const int r  = tid >> 3;
            const int c0 = (tid & 7) * 16;
            float xr[LAG];
            const size_t base = ((size_t)s * BATCH + (size_t)(bBase + r)) * LAG;
#pragma unroll
            for (int j = 0; j < LAG; ++j) xr[j] = inputs[base + j];
#pragma unroll
            for (int c = 0; c < 16; ++c) {
                float a = b_inp[c0 + c];
#pragma unroll
                for (int j = 0; j < LAG; ++j) a += xr[j] * W_inp[(c0 + c) * LAG + j];
                act0[r][c0 + c] = f2bf(a);
            }
        }
        __syncthreads();

        // ---- layer 0
        lstm_layer(act0, wpack, biasL[0], mt, hcBase, lane, c1, h1f);
        __syncthreads();   // everyone done reading act0 before h1 overwrite
#pragma unroll
        for (int t = 0; t < 2; ++t)
#pragma unroll
            for (int v = 0; v < 8; ++v) {
                const int row = mt * 16 + 8 * hi + v;    // C: M = v + 8*(lane/16)
                const int col = hcBase + t * 16 + ln16;  // C: N = lane%16
                const unsigned short hb = f2bf(h1f[t][v]);
                act1[row][col]     = hb;                 // layer-1 input
                act0[row][H + col] = hb;                 // next-step recurrent h1
            }
        __syncthreads();

        // ---- layer 1
        lstm_layer(act1, wpack + WPK_L, biasL[1], mt, hcBase, lane, c2, h2f);
        __syncthreads();   // everyone done reading act1 before h2 overwrite
#pragma unroll
        for (int t = 0; t < 2; ++t)
#pragma unroll
            for (int v = 0; v < 8; ++v) {
                const int row = mt * 16 + 8 * hi + v;
                const int col = hcBase + t * 16 + ln16;
                act1[row][H + col] = f2bf(h2f[t][v]);
            }
        __syncthreads();

        // ---- per-step head: y = h2 @ W_out^T + b_out  -> out[b][s]
        {
            const int r = tid >> 3, seg = tid & 7;
            float p = 0.0f;
#pragma unroll
            for (int k = 0; k < 16; ++k)
                p += bf2f(act1[r][H + seg * 16 + k]) * W_out[seg * 16 + k];
            p += __shfl_xor(p, 4, 32);
            p += __shfl_xor(p, 2, 32);
            p += __shfl_xor(p, 1, 32);
            if (seg == 0)
                out[(size_t)(bBase + r) * (STEPS + FH) + s] = p + b_out[0];
        }
        __syncthreads();
    }

    // ---- forecast: h2 @ W_fc^T + b_fc -> out[b][10..14]
    if (tid < BT * FH) {
        const int r = tid / FH, j = tid % FH;
        float a = b_fc[j];
        for (int k = 0; k < H; ++k)
            a += bf2f(act1[r][H + k]) * W_fc[j * H + k];
        out[(size_t)(bBase + r) * (STEPS + FH) + STEPS + j] = a;
    }

    // ---- h_stack [2,B,H]
    const size_t hBase = (size_t)BATCH * (STEPS + FH);
    for (int idx = tid; idx < 2 * BT * H; idx += 256) {
        const int layer = idx >> 12;
        const int rem   = idx & 4095;
        const int r = rem >> 7, k = rem & 127;
        const unsigned short hb = layer ? act1[r][H + k] : act0[r][H + k];
        out[hBase + (size_t)layer * BATCH * H + (size_t)(bBase + r) * H + k] = bf2f(hb);
    }

    // ---- c_stack [2,B,H] straight from register fragments
    const size_t cBase = hBase + (size_t)2 * BATCH * H;
#pragma unroll
    for (int t = 0; t < 2; ++t)
#pragma unroll
        for (int v = 0; v < 8; ++v) {
            const int row = mt * 16 + 8 * hi + v;
            const int col = hcBase + t * 16 + ln16;
            const size_t o = (size_t)(bBase + row) * H + col;
            out[cBase + o]                       = c1[t][v];
            out[cBase + (size_t)BATCH * H + o]   = c2[t][v];
        }
}

// ---------------------------------------------------------------------------
extern "C" void kernel_launch(void* const* d_in, const int* in_sizes, int n_in,
                              void* d_out, int out_size, void* d_ws, size_t ws_size,
                              hipStream_t stream) {
    const float* inputs = (const float*)d_in[0];
    const float* W_inp  = (const float*)d_in[1];
    const float* b_inp  = (const float*)d_in[2];
    const float* W_ih0  = (const float*)d_in[3];
    const float* W_hh0  = (const float*)d_in[4];
    const float* b_ih0  = (const float*)d_in[5];
    const float* b_hh0  = (const float*)d_in[6];
    const float* W_ih1  = (const float*)d_in[7];
    const float* W_hh1  = (const float*)d_in[8];
    const float* b_ih1  = (const float*)d_in[9];
    const float* b_hh1  = (const float*)d_in[10];
    const float* W_out  = (const float*)d_in[11];
    const float* b_out  = (const float*)d_in[12];
    const float* W_fc   = (const float*)d_in[13];
    const float* b_fc   = (const float*)d_in[14];

    unsigned short* wpack = (unsigned short*)d_ws;   // 512 KB bf16 packed weights
    float* out = (float*)d_out;

    pack_weights<<<(2 * WPK_L) / 256, 256, 0, stream>>>(W_ih0, W_hh0, W_ih1, W_hh1, wpack);
    lstm2_forward<<<BATCH / BT, 256, 0, stream>>>(inputs, W_inp, b_inp,
                                                  b_ih0, b_hh0, b_ih1, b_hh1,
                                                  W_out, b_out, W_fc, b_fc,
                                                  wpack, out);
}
